// MultiHeadAttn_5970004541835
// MI455X (gfx1250) — matrix operand order repaired and checked by reading, not compiled, not run on hardware
//
#include <hip/hip_runtime.h>

// Problem constants (match reference)
#define S_LEN   2048
#define DMODEL  1024
#define NHEAD   16
#define ADIM    64
#define BREF    2
#define MROWS   (BREF * S_LEN)       // 4096 rows for [B*S, D] GEMMs
#define NREL    33                   // 2*K_REL + 1
#define KITER   (S_LEN / 32)         // 64 key-pair iterations

typedef __attribute__((ext_vector_type(16))) __bf16 v16bf;
typedef __attribute__((ext_vector_type(8)))  __bf16 v8bf;
typedef __attribute__((ext_vector_type(4)))  __bf16 v4bf;
typedef __attribute__((ext_vector_type(8)))  float  v8f;
typedef __attribute__((ext_vector_type(4)))  float  v4f;
typedef __attribute__((ext_vector_type(4)))  int    v4i;

static __device__ __forceinline__ v8bf ld8(const __bf16* p) {
    return *reinterpret_cast<const v8bf*>(p);
}
static __device__ __forceinline__ v16bf cat16(v8bf a, v8bf b) {
    return __builtin_shufflevector(a, b, 0,1,2,3,4,5,6,7,8,9,10,11,12,13,14,15);
}
static __device__ __forceinline__ v8f wmma_bf16(v16bf a, v16bf b, v8f c) {
    // D = A(16x32 bf16) x B(32x16 bf16) + C(16x16 f32)
    return __builtin_amdgcn_wmma_f32_16x16x32_bf16(false, a, false, b, (short)0, c, false, false);
}
// Async DMA: 16 bytes/lane global -> LDS, tracked by ASYNCcnt.
static __device__ __forceinline__ void async_b128(unsigned lds_off, const void* gaddr) {
    asm volatile("global_load_async_to_lds_b128 %0, %1, off"
                 : : "v"(lds_off), "v"(gaddr) : "memory");
}
static __device__ __forceinline__ unsigned lds_lo(const void* p) {
    return (unsigned)(unsigned long long)p;   // low 32 bits of flat shared addr = LDS offset
}

// ---------------------------------------------------------------------------
// fp32 -> bf16 conversion, 4 elements/thread/step
// ---------------------------------------------------------------------------
__global__ __launch_bounds__(256) void cvt_bf16_kernel(const float* __restrict__ src,
                                                       __bf16* __restrict__ dst, int n4) {
    int i = blockIdx.x * blockDim.x + threadIdx.x;
    int stride = gridDim.x * blockDim.x;
    for (; i < n4; i += stride) {
        v4f x = reinterpret_cast<const v4f*>(src)[i];
        v4bf y;
        y.x = (__bf16)x.x; y.y = (__bf16)x.y; y.z = (__bf16)x.z; y.w = (__bf16)x.w;
        reinterpret_cast<v4bf*>(dst)[i] = y;
    }
}

// ---------------------------------------------------------------------------
// Pack int mask [B,S,S] into bitmask words [B,S,S/32] (bit j = key w*32+j)
// ---------------------------------------------------------------------------
__global__ __launch_bounds__(256) void pack_mask_kernel(const int* __restrict__ m,
                                                        unsigned* __restrict__ bits, int nwords) {
    int i = blockIdx.x * blockDim.x + threadIdx.x;
    int stride = gridDim.x * blockDim.x;
    for (; i < nwords; i += stride) {
        const v4i* base = reinterpret_cast<const v4i*>(m + (size_t)i * 32);
        unsigned w = 0;
#pragma unroll
        for (int j = 0; j < 8; ++j) {
            v4i q = base[j];
            w |= (q.x != 0 ? 1u : 0u) << (j * 4 + 0);
            w |= (q.y != 0 ? 1u : 0u) << (j * 4 + 1);
            w |= (q.z != 0 ? 1u : 0u) << (j * 4 + 2);
            w |= (q.w != 0 ? 1u : 0u) << (j * 4 + 3);
        }
        bits[i] = w;
    }
}

// ---------------------------------------------------------------------------
// GEMM: out[m,n] = sum_k X[m,k] * W[n,k]  (X: MxK bf16, W: NxK bf16, K=1024)
// Block = 8 waves sharing one 64-col W panel: the W k-slice (64x32 bf16, 4KB)
// is async-DMA'd into double-buffered LDS (1 async_b128/thread/tile).
// A-stream stays register-pipelined from global. One wave -> 16x64 tile.
// mode 0/1: bf16 [B,H,S,ADIM] | mode 2: bf16 [B,H,ADIM,S] | mode 3: f32 [M,DMODEL]
// ---------------------------------------------------------------------------
__global__ __launch_bounds__(256) void gemm_proj_kernel(const __bf16* __restrict__ X,
                                                        const __bf16* __restrict__ W,
                                                        void* __restrict__ out,
                                                        int M, int mode) {
    __shared__ __align__(16) __bf16 wlds[2][64 * 32];   // [n-local][k-local]

    const int K = DMODEL;
    const int tid  = threadIdx.x;
    const int lane = tid & 31;
    const int wave = tid >> 5;
    const int wid  = blockIdx.x * 8 + wave;
    const int mTiles = M >> 4;
    const int mt = wid % mTiles;
    const int ng = wid / mTiles;          // block-uniform (8 divides mTiles)
    const int m0 = mt << 4;
    const int n0 = ng << 6;

    const int aks  = (lane < 16) ? 0 : 16;
    const int bks  = (lane < 16) ? 0 : 16;
    const int ncol = lane & 15;

    // 256 threads x 16B = one 4KB W slice per async instruction set (cnt 1/wave)
    auto issue_w = [&](int buf, int kk) {
        const char* g = (const char*)W +
            ((size_t)(n0 + (tid >> 2)) * K + kk) * 2 + (size_t)(tid & 3) * 16;
        async_b128(lds_lo(&wlds[buf][0]) + (unsigned)tid * 16u, g);
    };

    const __bf16* xrow = X + (size_t)(m0 + (lane & 15)) * K;
    auto loadA = [&](int kk) {
        return cat16(ld8(xrow + kk + aks), ld8(xrow + kk + aks + 8));
    };

    v8f acc[4] = {};
    issue_w(0, 0);
    issue_w(1, 32);
    v16bf ca = loadA(0);

    const int NIT = K / 32;               // 32 k-steps
    for (int it = 0; it < NIT; ++it) {
        const int cur = it & 1;
        v16bf na = ca;
        if (it + 1 < NIT) na = loadA((it + 1) * 32);   // A for next step in flight
        if (it + 1 < NIT) asm volatile("s_wait_asynccnt 0x1" ::: "memory");
        else              asm volatile("s_wait_asynccnt 0x0" ::: "memory");
        __syncthreads();                               // W slice visible block-wide
#pragma unroll
        for (int j = 0; j < 4; ++j) {
            const __bf16* wr = &wlds[cur][(j * 16 + ncol) * 32 + bks];
            v16bf bf = cat16(*(const v8bf*)wr, *(const v8bf*)(wr + 8));
            acc[j] = wmma_bf16(ca, bf, acc[j]);
        }
        __syncthreads();                               // all waves done with slice
        if (it + 2 < NIT) issue_w(cur, (it + 2) * 32);
        ca = na;
    }

    const int rhalf = (lane < 16) ? 0 : 8;
#pragma unroll
    for (int r = 0; r < 8; ++r) {
        int m = m0 + r + rhalf;
        int b = m >> 11;
        int s = m & (S_LEN - 1);
#pragma unroll
        for (int j = 0; j < 4; ++j) {
            int n = n0 + j * 16 + ncol;
            float v = acc[j][r];
            if (mode == 3) {
                ((float*)out)[(size_t)m * DMODEL + n] = v;
            } else {
                int h = n >> 6;
                int d = n & 63;
                size_t addr;
                if (mode == 2)
                    addr = (((size_t)b * NHEAD + h) * ADIM + d) * S_LEN + s;
                else
                    addr = (((size_t)b * NHEAD + h) * S_LEN + s) * ADIM + d;
                ((__bf16*)out)[addr] = (__bf16)v;
            }
        }
    }
}

// ---------------------------------------------------------------------------
// qe[row, r] = sum_d q_bf16[row, d] * rel_emb[r, d]
// ---------------------------------------------------------------------------
__global__ __launch_bounds__(256) void qe_kernel(const __bf16* __restrict__ qb,
                                                 const float* __restrict__ rel,
                                                 float* __restrict__ qe, int nrows) {
    __shared__ float relS[NREL * ADIM];
    for (int i = threadIdx.x; i < NREL * ADIM; i += 256) relS[i] = rel[i];
    __syncthreads();

    int row = blockIdx.x * 256 + threadIdx.x;
    if (row >= nrows) return;
    const __bf16* qr = qb + (size_t)row * ADIM;
    float q[ADIM];
#pragma unroll
    for (int d = 0; d < ADIM; ++d) q[d] = (float)qr[d];
    for (int r = 0; r < NREL; ++r) {
        float sum = 0.f;
#pragma unroll
        for (int d = 0; d < ADIM; ++d) sum += q[d] * relS[r * ADIM + d];
        qe[(size_t)row * NREL + r] = sum;
    }
}

// ---------------------------------------------------------------------------
// Fused flash-style attention, ReLU^2 normalizer.
// Block = 4 waves = 4 q-tiles of the SAME (b,h); K/V tiles async-DMA'd into
// double-buffered LDS (4 async_b128/wave/tile) and shared by all 4 waves.
// ---------------------------------------------------------------------------
__global__ __launch_bounds__(128) void attn_kernel(const __bf16* __restrict__ qb,
                                                   const __bf16* __restrict__ kb,
                                                   const __bf16* __restrict__ vt,
                                                   const float* __restrict__ qe,
                                                   const unsigned* __restrict__ mbits,
                                                   const float* __restrict__ spb,
                                                   __bf16* __restrict__ ao) {
    __shared__ __align__(16) __bf16 klds[2][32 * ADIM];   // [keys][d]
    __shared__ __align__(16) __bf16 vlds[2][ADIM * 32];   // [d][keys]
    __shared__ __align__(16) __bf16 tls[4][16 * 32];      // per-wave t relayout
    __shared__ float qes[4][16 * NREL];

    const int tid  = threadIdx.x;
    const int lane = tid & 31;
    const int wave = tid >> 5;
    const int wid  = blockIdx.x * 4 + wave;
    const int qt   = wid & 127;
    const int bh   = wid >> 7;            // uniform across the block
    const int b    = bh >> 4;
    const int hh   = bh & 15;
    const int q0   = qt * 16;

    const __bf16* qbase = qb + (size_t)bh * S_LEN * ADIM;
    const __bf16* kbase = kb + (size_t)bh * S_LEN * ADIM;
    const __bf16* vbase = vt + (size_t)bh * ADIM * S_LEN;
    const float*  qeb   = qe + ((size_t)bh * S_LEN + q0) * NREL;
    const unsigned* mbase = mbits + ((size_t)b * S_LEN + q0) * (S_LEN / 32);
    const float   sbias = spb[0];

    auto issue_tile = [&](int buf, int k0) {
        const char* kg = (const char*)(kbase + (size_t)k0 * ADIM);
        unsigned kl = lds_lo(&klds[buf][0]);
#pragma unroll
        for (int c = 0; c < 2; ++c) {
            unsigned off = (unsigned)(tid + c * 128) * 16u;
            async_b128(kl + off, kg + off);
        }
        const char* vg = (const char*)vbase + (size_t)k0 * 2;
        unsigned vl = lds_lo(&vlds[buf][0]);
#pragma unroll
        for (int c = 0; c < 2; ++c) {
            int chunk = tid + c * 128;                  // 256 chunks of 16B
            const char* g = vg + (size_t)(chunk >> 2) * (S_LEN * 2) + (size_t)(chunk & 3) * 16;
            async_b128(vl + (unsigned)chunk * 16u, g);
        }
    };

    issue_tile(0, 0);
    issue_tile(1, 32);

    for (int i = lane; i < 16 * NREL; i += 32) qes[wave][i] = qeb[i];

    const int aks   = (lane < 16) ? 0 : 16;
    const int bko   = (lane < 16) ? 0 : 16;
    const int ncol  = lane & 15;
    const int rhalf = (lane < 16) ? 0 : 8;
    const __bf16* qrow = qbase + (size_t)(q0 + (lane & 15)) * ADIM;
    v16bf qf0 = cat16(ld8(qrow + aks),      ld8(qrow + aks + 8));
    v16bf qf1 = cat16(ld8(qrow + 32 + aks), ld8(qrow + 32 + aks + 8));

    v8f   o[4]    = {};
    float dsum[8] = {};

    for (int it = 0; it < KITER; ++it) {
        const int cur = it & 1;
        unsigned mrow[8];
#pragma unroll
        for (int r = 0; r < 8; ++r) mrow[r] = mbase[(size_t)(r + rhalf) * (S_LEN / 32) + it];

        if (it + 1 < KITER) asm volatile("s_wait_asynccnt 0x4" ::: "memory");
        else                asm volatile("s_wait_asynccnt 0x0" ::: "memory");
        __syncthreads();                                  // staged K/V visible

#pragma unroll
        for (int sub = 0; sub < 2; ++sub) {
            const __bf16* kro = &klds[cur][(sub * 16 + ncol) * ADIM + bko];
            v16bf kf0 = cat16(*(const v8bf*)kro,        *(const v8bf*)(kro + 8));
            v16bf kf1 = cat16(*(const v8bf*)(kro + 32), *(const v8bf*)(kro + 40));
            v8f s = {};
            s = wmma_bf16(qf0, kf0, s);
            s = wmma_bf16(qf1, kf1, s);
#pragma unroll
            for (int r = 0; r < 8; ++r) {
                int m = r + rhalf;
                int kidx = it * 32 + sub * 16 + ncol;
                int rel = kidx - (q0 + m);
                rel = (rel < -16) ? -16 : (rel > 16 ? 16 : rel);
                float sv = (s[r] + qes[wave][m * NREL + rel + 16]) * 0.125f;
                if ((mrow[r] >> (sub * 16 + ncol)) & 1u) sv = -1e9f;
                float t = sv + sbias;
                t = (t > 0.f) ? t * t : 0.f;
                dsum[r] += t;
                tls[wave][m * 32 + sub * 16 + ncol] = (__bf16)t;
            }
        }
        __syncthreads();                                  // t tile ready
        const __bf16* tp = &tls[wave][(lane & 15) * 32 + aks];
        v16bf tf = cat16(*(const v8bf*)tp, *(const v8bf*)(tp + 8));
#pragma unroll
        for (int j = 0; j < 4; ++j) {
            const __bf16* vro = &vlds[cur][(j * 16 + ncol) * 32 + bko];
            v16bf vf = cat16(*(const v8bf*)vro, *(const v8bf*)(vro + 8));
            o[j] = wmma_bf16(tf, vf, o[j]);
        }
        __syncthreads();                                  // all waves done with cur
        if (it + 2 < KITER) issue_tile(cur, (it + 2) * 32);
    }

#pragma unroll
    for (int r = 0; r < 8; ++r) {
        float v = dsum[r];
        v += __shfl_xor(v, 1, 32);
        v += __shfl_xor(v, 2, 32);
        v += __shfl_xor(v, 4, 32);
        v += __shfl_xor(v, 8, 32);
        dsum[r] = v;
    }
#pragma unroll
    for (int r = 0; r < 8; ++r) {
        int m = r + rhalf;
        float inv = 1.0f / (dsum[r] + 1e-9f);
        size_t base = ((size_t)b * S_LEN + q0 + m) * DMODEL + hh * ADIM;
#pragma unroll
        for (int j = 0; j < 4; ++j)
            ao[base + j * 16 + ncol] = (__bf16)(o[j][r] * inv);
    }
}

// ---------------------------------------------------------------------------
extern "C" void kernel_launch(void* const* d_in, const int* in_sizes, int n_in,
                              void* d_out, int out_size, void* d_ws, size_t ws_size,
                              hipStream_t stream) {
    const float* iQ   = (const float*)d_in[0];
    const float* iK   = (const float*)d_in[1];
    const float* iV   = (const float*)d_in[2];
    const int*   mask = (const int*)  d_in[3];
    const float* Wq   = (const float*)d_in[4];
    const float* Wk   = (const float*)d_in[5];
    const float* Wv   = (const float*)d_in[6];
    const float* Wo   = (const float*)d_in[7];
    const float* rel  = (const float*)d_in[8];
    const float* spb  = (const float*)d_in[9];
    float* out = (float*)d_out;

    char* ws = (char*)d_ws;
    size_t off = 0;
    auto alloc = [&](size_t bytes) -> void* {
        void* p = ws + off;
        off += (bytes + 255) & ~(size_t)255;
        return p;
    };

    const size_t XE = (size_t)MROWS * DMODEL;
    const size_t WE = (size_t)DMODEL * DMODEL;
    const size_t HE = (size_t)BREF * NHEAD * S_LEN * ADIM;
    const int    MW = BREF * S_LEN * (S_LEN / 32);      // mask words

    __bf16* xq  = (__bf16*)alloc(XE * 2);
    __bf16* xk  = (__bf16*)alloc(XE * 2);
    __bf16* xv  = (__bf16*)alloc(XE * 2);
    __bf16* wqb = (__bf16*)alloc(WE * 2);
    __bf16* wkb = (__bf16*)alloc(WE * 2);
    __bf16* wvb = (__bf16*)alloc(WE * 2);
    __bf16* wob = (__bf16*)alloc(WE * 2);
    __bf16* qbf = (__bf16*)alloc(HE * 2);               // [B,H,S,ADIM]
    __bf16* kbf = (__bf16*)alloc(HE * 2);               // [B,H,S,ADIM]
    __bf16* vtb = (__bf16*)alloc(HE * 2);               // [B,H,ADIM,S]
    float*  qe  = (float*) alloc((size_t)BREF * NHEAD * S_LEN * NREL * 4);
    unsigned* mb = (unsigned*)alloc((size_t)MW * 4);
    __bf16* ao  = xq;                                   // alias: xq dead after Q proj

    cvt_bf16_kernel<<<2048, 256, 0, stream>>>(iQ, xq, (int)(XE / 4));
    cvt_bf16_kernel<<<2048, 256, 0, stream>>>(iK, xk, (int)(XE / 4));
    cvt_bf16_kernel<<<2048, 256, 0, stream>>>(iV, xv, (int)(XE / 4));
    cvt_bf16_kernel<<<512, 256, 0, stream>>>(Wq, wqb, (int)(WE / 4));
    cvt_bf16_kernel<<<512, 256, 0, stream>>>(Wk, wkb, (int)(WE / 4));
    cvt_bf16_kernel<<<512, 256, 0, stream>>>(Wv, wvb, (int)(WE / 4));
    cvt_bf16_kernel<<<512, 256, 0, stream>>>(Wo, wob, (int)(WE / 4));
    pack_mask_kernel<<<1024, 256, 0, stream>>>(mask, mb, MW);

    gemm_proj_kernel<<<512, 256, 0, stream>>>(xq, wqb, qbf, MROWS, 0);
    gemm_proj_kernel<<<512, 256, 0, stream>>>(xk, wkb, kbf, MROWS, 1);
    gemm_proj_kernel<<<512, 256, 0, stream>>>(xv, wvb, vtb, MROWS, 2);

    int nrows = BREF * NHEAD * S_LEN;
    qe_kernel<<<(nrows + 255) / 256, 256, 0, stream>>>(qbf, rel, qe, nrows);

    attn_kernel<<<1024, 128, 0, stream>>>(qbf, kbf, vtb, qe, mb, spb, ao);

    gemm_proj_kernel<<<512, 256, 0, stream>>>(ao, wob, out, MROWS, 3);
}
